// LSTMRegressor_19181323944051
// MI455X (gfx1250) — matrix operand order repaired and checked by reading, not compiled, not run on hardware
//
#include <hip/hip_runtime.h>

// 2-layer LSTM (H=64), B=2048, T=512, fused per-timestep, f16 WMMA + f32 accum.
// Each wave32 owns a 16-row batch tile; weights LDS-resident pre-swizzled into
// B-fragment order; h state round-trips through a wave-private LDS staging tile.
// Gate math processed in 4 independent q-groups to keep accumulator VGPR
// pressure low (4 live C tiles instead of 16).

#define HDIM 64
#define TSTEPS 512
#define NT 16          // 4H/16 gate column tiles
#define KF0 2          // K fragments layer0 (64/32)
#define KF1 4          // K fragments layer1 concat (128/32)
#define NWAVES 2
#define ROWS_PER_BLOCK (NWAVES * 16)

typedef _Float16 f16;
typedef _Float16 v16h __attribute__((ext_vector_type(16)));
typedef _Float16 v8h  __attribute__((ext_vector_type(8)));
typedef float    v8f  __attribute__((ext_vector_type(8)));

__device__ __forceinline__ v8f wmma16(v16h a, v16h b, v8f c) {
  // D = A(16x32 f16) * B(32x16 f16) + C(16x16 f32)
  return __builtin_amdgcn_wmma_f32_16x16x32_f16(false, a, false, b, (short)0, c,
                                                false, false);
}

#if __has_builtin(__builtin_amdgcn_tanhf)
// CDNA5 hardware transcendental tanh (TRANS class -> co-executes with WMMA)
__device__ __forceinline__ float tanh_a(float x) { return __builtin_amdgcn_tanhf(x); }
__device__ __forceinline__ float sigm(float x) {
  return fmaf(0.5f, tanh_a(0.5f * x), 0.5f);
}
#else
__device__ __forceinline__ float tanh_a(float x) {
  x = fminf(fmaxf(x, -15.0f), 15.0f);
  float e = __expf(2.0f * x);
  return (e - 1.0f) / (e + 1.0f);
}
__device__ __forceinline__ float sigm(float x) { return 1.0f / (1.0f + __expf(-x)); }
#endif

// A-fragment (16x32 f16) from a row-major [16][64] f16 staging tile.
// Lane: m = lane%16, half = lane/16. Element j -> K=kf*32+16*half+j:
// two contiguous 16B chunks.
__device__ __forceinline__ v16h load_afrag(const f16* stage, int nl, int hf, int kf) {
  union { v16h v; v8h h[2]; } u;
  const f16* p = stage + nl * HDIM + kf * 32 + hf * 16;
  u.h[0] = *(const v8h*)(p);
  u.h[1] = *(const v8h*)(p + 8);
  return u.v;
}

// B-fragment (32x16 f16) from pre-swizzled LDS: 32 contiguous bytes per lane.
__device__ __forceinline__ v16h load_bfrag(const f16* wb, int lane, int nt, int kf, int KF) {
  return *(const v16h*)(wb + (((nt * KF) + kf) * 32 + lane) * 16);
}

__global__ __launch_bounds__(NWAVES * 32) void lstm2_fused_kernel(
    const float* __restrict__ x,
    const float* __restrict__ W_ih0, const float* __restrict__ W_hh0,
    const float* __restrict__ b_ih0, const float* __restrict__ b_hh0,
    const float* __restrict__ W_ih1, const float* __restrict__ W_hh1,
    const float* __restrict__ b_ih1, const float* __restrict__ b_hh1,
    const float* __restrict__ fc_W,  const float* __restrict__ fc_b,
    float* __restrict__ out)
{
  extern __shared__ char smem_raw[];
  f16* wbuf0  = (f16*)smem_raw;                 // 16*2*512  f16 = 32KB
  f16* wbuf1  = wbuf0 + NT * KF0 * 512;         // 16*4*512  f16 = 64KB
  f16* xtiles = wbuf1 + NT * KF1 * 512;         // NWAVES*512*16 f16
  f16* stages = xtiles + NWAVES * TSTEPS * 16;  // NWAVES*2048 f16 (h1 + h2)

  const int tid  = threadIdx.x;
  const int nthr = blockDim.x;
  const int wave = tid >> 5;
  const int lane = tid & 31;
  const int hf   = lane >> 4;   // half of wave
  const int nl   = lane & 15;   // column within tile / row within tile

  // ---- cooperative LDS init: swizzle weights into B-fragment order ----
  // slot(nt,kf,lane,j) holds Wsrc[n = nt*16 + lane%16][k = kf*32 + 16*(lane/16) + j]
  for (int idx = tid; idx < NT * KF0 * 512; idx += nthr) {
    int j  = idx & 15;
    int l  = (idx >> 4) & 31;
    int kf = (idx >> 9) & 1;
    int nt = idx >> 10;
    int n  = nt * 16 + (l & 15);
    int k  = kf * 32 + (l >> 4) * 16 + j;
    wbuf0[idx] = (f16)W_hh0[n * HDIM + k];
  }
  // layer1: K = [ h2 (W_hh1) ; h1_t (W_ih1) ], 128 wide
  for (int idx = tid; idx < NT * KF1 * 512; idx += nthr) {
    int j  = idx & 15;
    int l  = (idx >> 4) & 31;
    int kf = (idx >> 9) & 3;
    int nt = idx >> 11;
    int n  = nt * 16 + (l & 15);
    int k  = kf * 32 + (l >> 4) * 16 + j;
    float v = (k < HDIM) ? W_hh1[n * HDIM + k] : W_ih1[n * HDIM + (k - HDIM)];
    wbuf1[idx] = (f16)v;
  }
  // x tiles: per wave, transposed [t][16] f16
  {
    const int rowbase = blockIdx.x * ROWS_PER_BLOCK;
    for (int idx = tid; idx < NWAVES * TSTEPS * 16; idx += nthr) {
      int m = idx & 15;
      int t = (idx >> 4) & (TSTEPS - 1);
      int w = idx >> 13;  // TSTEPS*16 == 8192
      xtiles[idx] = (f16)x[(size_t)(rowbase + w * 16 + m) * TSTEPS + t];
    }
  }
  for (int idx = tid; idx < NWAVES * 2048; idx += nthr) stages[idx] = (f16)0.0f;
  __syncthreads();

  f16* xt  = xtiles + wave * (TSTEPS * 16);
  f16* sh1 = stages + wave * 2048;   // [16][64] f16, h1 state
  f16* sh2 = sh1 + 1024;             // [16][64] f16, h2 state

  // per-lane constants: lane owns gate column n = nt*16 + nl of each tile
  float w0v[NT], bb0[NT], bb1[NT];
  #pragma unroll
  for (int nt = 0; nt < NT; ++nt) {
    int n = nt * 16 + nl;
    w0v[nt] = W_ih0[n];                 // [4H,1]
    bb0[nt] = b_ih0[n] + b_hh0[n];
    bb1[nt] = b_ih1[n] + b_hh1[n];
  }
  float fcw[4];
  #pragma unroll
  for (int q = 0; q < 4; ++q) fcw[q] = fc_W[q * 16 + nl];

  v8f c1[4], c2[4];
  #pragma unroll
  for (int q = 0; q < 4; ++q) {
    #pragma unroll
    for (int r = 0; r < 8; ++r) { c1[q][r] = 0.f; c2[q][r] = 0.f; }
  }

  #pragma unroll 1
  for (int t = 0; t < TSTEPS; ++t) {
    // ===================== layer 0 =====================
    v16h a0 = load_afrag(sh1, nl, hf, 0);
    v16h a1 = load_afrag(sh1, nl, hf, 1);
    v8h  xv = *(const v8h*)(xt + t * 16 + hf * 8);   // x for this lane's 8 rows
    // process gate tiles as 4 independent groups {q, q+4, q+8, q+12}
    #pragma unroll
    for (int q = 0; q < 4; ++q) {
      v8f gi, gf, gg, go;
      #pragma unroll
      for (int r = 0; r < 8; ++r) {
        float xr = (float)xv[r];
        gi[r] = fmaf(xr, w0v[q],      bb0[q]);
        gf[r] = fmaf(xr, w0v[4 + q],  bb0[4 + q]);
        gg[r] = fmaf(xr, w0v[8 + q],  bb0[8 + q]);
        go[r] = fmaf(xr, w0v[12 + q], bb0[12 + q]);
      }
      gi = wmma16(a0, load_bfrag(wbuf0, lane, q,      0, KF0), gi);
      gi = wmma16(a1, load_bfrag(wbuf0, lane, q,      1, KF0), gi);
      gf = wmma16(a0, load_bfrag(wbuf0, lane, 4 + q,  0, KF0), gf);
      gf = wmma16(a1, load_bfrag(wbuf0, lane, 4 + q,  1, KF0), gf);
      gg = wmma16(a0, load_bfrag(wbuf0, lane, 8 + q,  0, KF0), gg);
      gg = wmma16(a1, load_bfrag(wbuf0, lane, 8 + q,  1, KF0), gg);
      go = wmma16(a0, load_bfrag(wbuf0, lane, 12 + q, 0, KF0), go);
      go = wmma16(a1, load_bfrag(wbuf0, lane, 12 + q, 1, KF0), go);
      #pragma unroll
      for (int r = 0; r < 8; ++r) {
        float cv = sigm(gf[r]) * c1[q][r] + sigm(gi[r]) * tanh_a(gg[r]);
        c1[q][r] = cv;
        float hv = sigm(go[r]) * tanh_a(cv);
        sh1[(r + hf * 8) * HDIM + q * 16 + nl] = (f16)hv;  // C-layout -> staging
      }
    }
    asm volatile("s_wait_dscnt 0x0" ::: "memory");  // wave-local LDS visibility

    // ============ layer 1: K = [h2 ; h1_t] (128) ============
    v16h f0 = load_afrag(sh2, nl, hf, 0);
    v16h f1 = load_afrag(sh2, nl, hf, 1);
    v16h f2 = load_afrag(sh1, nl, hf, 0);
    v16h f3 = load_afrag(sh1, nl, hf, 1);
    #pragma unroll
    for (int q = 0; q < 4; ++q) {
      v8f gi, gf, gg, go;
      #pragma unroll
      for (int r = 0; r < 8; ++r) {
        gi[r] = bb1[q];
        gf[r] = bb1[4 + q];
        gg[r] = bb1[8 + q];
        go[r] = bb1[12 + q];
      }
      #pragma unroll
      for (int kf = 0; kf < KF1; ++kf) {
        v16h af = (kf == 0) ? f0 : (kf == 1) ? f1 : (kf == 2) ? f2 : f3;
        gi = wmma16(af, load_bfrag(wbuf1, lane, q,      kf, KF1), gi);
        gf = wmma16(af, load_bfrag(wbuf1, lane, 4 + q,  kf, KF1), gf);
        gg = wmma16(af, load_bfrag(wbuf1, lane, 8 + q,  kf, KF1), gg);
        go = wmma16(af, load_bfrag(wbuf1, lane, 12 + q, kf, KF1), go);
      }
      #pragma unroll
      for (int r = 0; r < 8; ++r) {
        float cv = sigm(gf[r]) * c2[q][r] + sigm(gi[r]) * tanh_a(gg[r]);
        c2[q][r] = cv;
        float hv = sigm(go[r]) * tanh_a(cv);
        sh2[(r + hf * 8) * HDIM + q * 16 + nl] = (f16)hv;
      }
    }
    asm volatile("s_wait_dscnt 0x0" ::: "memory");
  }

  // ===== final FC: out[m] = sum_n h2_last[m,n] * fc_W[n] + fc_b =====
  // h2_last lives in sh2 (f16); reduce via [16][16] f32 scratch in sh1.
  float* red = (float*)sh1;
  #pragma unroll
  for (int r = 0; r < 8; ++r) {
    float a = 0.f;
    #pragma unroll
    for (int q = 0; q < 4; ++q)
      a = fmaf((float)sh2[(r + hf * 8) * HDIM + q * 16 + nl], fcw[q], a);
    red[(r + hf * 8) * 16 + nl] = a;  // partial for (row m, col-chunk nl)
  }
  asm volatile("s_wait_dscnt 0x0" ::: "memory");
  if (lane < 16) {
    float a = fc_b[0];
    #pragma unroll
    for (int n = 0; n < 16; ++n) a += red[lane * 16 + n];
    out[blockIdx.x * ROWS_PER_BLOCK + wave * 16 + lane] = a;
  }
}

extern "C" void kernel_launch(void* const* d_in, const int* in_sizes, int n_in,
                              void* d_out, int out_size, void* d_ws, size_t ws_size,
                              hipStream_t stream) {
  const float* x     = (const float*)d_in[0];
  const float* W_ih0 = (const float*)d_in[1];
  const float* W_hh0 = (const float*)d_in[2];
  const float* b_ih0 = (const float*)d_in[3];
  const float* b_hh0 = (const float*)d_in[4];
  const float* W_ih1 = (const float*)d_in[5];
  const float* W_hh1 = (const float*)d_in[6];
  const float* b_ih1 = (const float*)d_in[7];
  const float* b_hh1 = (const float*)d_in[8];
  const float* fc_W  = (const float*)d_in[9];
  const float* fc_b  = (const float*)d_in[10];
  float* out = (float*)d_out;

  int B = in_sizes[0] / TSTEPS;            // 2048
  int nblocks = B / ROWS_PER_BLOCK;        // 64 WGs x 2 waves = 128 indep. tiles
  size_t shmem = (size_t)(NT * KF0 * 512 + NT * KF1 * 512 +
                          NWAVES * TSTEPS * 16 + NWAVES * 2048) * sizeof(f16);
  lstm2_fused_kernel<<<nblocks, NWAVES * 32, shmem, stream>>>(
      x, W_ih0, W_hh0, b_ih0, b_hh0, W_ih1, W_hh1, b_ih1, b_hh1, fc_W, fc_b, out);
}
